// TreeConv1d_5317169513228
// MI455X (gfx1250) — matrix operand order repaired and checked by reading, not compiled, not run on hardware
//
#include <hip/hip_runtime.h>

typedef __bf16 bf16_t;
typedef __attribute__((ext_vector_type(16))) __bf16 v16bf;
typedef __attribute__((ext_vector_type(8)))  float  v8f;

#define B_DIM   64
#define C_IN    256
#define C_OUT   256
#define N_NODES 2048
#define M_NODES 2048
#define K_TOT   768            // C_IN * 3
#define M_OUT   (M_NODES + 1)  // +1 dummy zero column

#define MT      128            // m-tile per workgroup (halves W re-read traffic)
#define NT      (MT / 16)      // 8 n-fragments per wave
#define KS      96             // K per stage = lcm(32, 3); 32 channels/stage
#define NSTAGE  (K_TOT / KS)   // 8
#define SBS     104            // sB row stride: 13 x 16B -> conflict-free b128
#define SBN     (MT * SBS)     // elements per sB buffer (26624 B)

// --- prep: convert W [O][C][3] f32 -> bf16 [O][K=768] (same flat order) ---
__global__ __launch_bounds__(256)
void cvt_w_kernel(const float* __restrict__ W, bf16_t* __restrict__ wb) {
    int t = blockIdx.x * 256 + threadIdx.x;      // < 196608
    wb[t] = (bf16_t)W[t];
}

// --- prep: zero column m=0 of feats; copy indexes (int bits) to output tail ---
__global__ __launch_bounds__(256)
void init_out_kernel(const int* __restrict__ idx, float* __restrict__ out) {
    int t = blockIdx.x * 256 + threadIdx.x;
    const int zeroN = B_DIM * C_OUT;             // 16384
    const int idxN  = B_DIM * 3 * M_NODES;       // 393216
    if (t < zeroN) out[(size_t)t * M_OUT] = 0.0f;
    int t2 = t - zeroN;
    if (t2 >= 0 && t2 < idxN) {
        ((int*)out)[(size_t)B_DIM * C_OUT * M_OUT + t2] = idx[t2];
    }
}

struct Frag32B { uint4 lo, hi; };

// 32B fragment load (global or LDS), 16B-aligned halves -> b128 pairs
__device__ __forceinline__ v16bf load_frag(const bf16_t* p) {
    Frag32B f;
    f.lo = *reinterpret_cast<const uint4*>(p);
    f.hi = *reinterpret_cast<const uint4*>(p + 8);
    return __builtin_bit_cast(v16bf, f);
}

// gather 24 f32 (8 channels x 3 children); channel = cbase + q3
__device__ __forceinline__ void gather_load(const float* __restrict__ dslice,
                                            int cbase,
                                            int n0, int n1, int n2,
                                            float gv[24]) {
#pragma unroll
    for (int q3 = 0; q3 < 8; ++q3) {
        const float* p = dslice + (size_t)(cbase + q3) * N_NODES;
        gv[q3 * 3 + 0] = p[n0];
        gv[q3 * 3 + 1] = p[n1];
        gv[q3 * 3 + 2] = p[n2];
    }
}

__device__ __forceinline__ void gather_store(bf16_t* __restrict__ dst,
                                             const float gv[24]) {
#pragma unroll
    for (int i = 0; i < 24; ++i) dst[i] = (bf16_t)gv[i];
}

// --- main: per (batch, m-tile) gathered GEMM 256x128x768, pipelined bf16 WMMA ---
__global__ __launch_bounds__(256)
void tree_conv_wmma_kernel(const float* __restrict__ data,
                           const int*   __restrict__ indexes,
                           const bf16_t* __restrict__ wb,
                           const float* __restrict__ bias,
                           float* __restrict__ out) {
    __shared__ __align__(16) bf16_t sB[2 * SBN];   // double-buffered gathered tile

    const int tid  = threadIdx.x;
    const int b    = blockIdx.y;
    const int m0   = blockIdx.x * MT;

    const int lane  = tid & 31;
    const int wv    = tid >> 5;          // 8 waves
    const int ln16  = lane & 15;
    const int lh    = lane >> 4;
    const int obase = wv * 32;           // wave's 32 output rows (disjoint in O)

    // gather identity: 2 threads per m-column, each owns 48 of the 96 K-slots
    // kloc = gq*48 + 3*q3 + kk  ->  c_local = gq*16 + q3, kk constant
    const int gm = tid >> 1;             // m within tile (0..127)
    const int gq = tid & 1;              // half of the K-stage
    const int ib = b * (3 * M_NODES) + 3 * (m0 + gm);
    const int n0 = indexes[ib + 0];
    const int n1 = indexes[ib + 1];
    const int n2 = indexes[ib + 2];
    bf16_t* const sRow0 = &sB[gm * SBS + gq * 48];          // buffer 0 slot
    const int     c16   = gq * 16;                          // channel sub-base

    const float* dslice = data + (size_t)b * C_IN * N_NODES;   // L2-resident

    // accumulators pre-loaded with bias (v8f rows are contiguous in o)
    v8f acc[2][NT];
#pragma unroll
    for (int oi = 0; oi < 2; ++oi) {
        v8f binit = *reinterpret_cast<const v8f*>(bias + obase + oi * 16 + 8 * lh);
#pragma unroll
        for (int ni = 0; ni < NT; ++ni) acc[oi][ni] = binit;
    }

    // one 32-wide kc chunk of WMMAs against the current sB buffer
    auto compute_kc = [&](const bf16_t* __restrict__ wbase,
                          const bf16_t* __restrict__ sBcur, int kc) {
        v16bf af[2];
#pragma unroll
        for (int oi = 0; oi < 2; ++oi) {
            int r = obase + oi * 16 + ln16;
            af[oi] = load_frag(wbase + (size_t)r * K_TOT + kc + 16 * lh);
        }
#pragma unroll
        for (int ni = 0; ni < NT; ++ni) {
            int n = ni * 16 + ln16;
            v16bf bfr = load_frag(sBcur + n * SBS + kc + 16 * lh);
#pragma unroll
            for (int oi = 0; oi < 2; ++oi)
                acc[oi][ni] = __builtin_amdgcn_wmma_f32_16x16x32_bf16(
                    false, af[oi], false, bfr,
                    (short)0, acc[oi][ni], false, false);
        }
    };

    // prologue: gather stage 0 (both halves) into buffer 0
    {
        float gv[24];
        gather_load(dslice, c16, n0, n1, n2, gv);
        gather_store(sRow0, gv);
        gather_load(dslice, c16 + 8, n0, n1, n2, gv);
        gather_store(sRow0 + 24, gv);
    }
    __syncthreads();

    for (int s = 0; s < NSTAGE; ++s) {
        const int cur = s & 1;
        const bf16_t* sBcur = &sB[cur * SBN];
        bf16_t* sNext = sRow0 + (1 - cur) * SBN;
        const bf16_t* wbase = wb + (size_t)s * KS;
        const int c0n = (s + 1) * 32;                // next stage's channel base
        const bool more = (s + 1 < NSTAGE);

        // half A of next gather flies under kc=0
        float gv[24];
        if (more) gather_load(dslice, c0n + c16, n0, n1, n2, gv);
        compute_kc(wbase, sBcur, 0);
        if (more) gather_store(sNext, gv);

        // half B flies under kc=32
        if (more) gather_load(dslice, c0n + c16 + 8, n0, n1, n2, gv);
        compute_kc(wbase, sBcur, 32);
        if (more) gather_store(sNext + 24, gv);

        compute_kc(wbase, sBcur, 64);
        __syncthreads();
    }

    // epilogue: D element g -> row g + 8*lh, col ln16; non-temporal stores
    float* outb = out + (size_t)b * C_OUT * M_OUT;
#pragma unroll
    for (int oi = 0; oi < 2; ++oi) {
#pragma unroll
        for (int ni = 0; ni < NT; ++ni) {
#pragma unroll
            for (int g = 0; g < 8; ++g) {
                int o = obase + oi * 16 + g + (lh << 3);
                int m = m0 + ni * 16 + ln16;
                __builtin_nontemporal_store(acc[oi][ni][g],
                                            &outb[(size_t)o * M_OUT + 1 + m]);
            }
        }
    }
}

extern "C" void kernel_launch(void* const* d_in, const int* in_sizes, int n_in,
                              void* d_out, int out_size, void* d_ws, size_t ws_size,
                              hipStream_t stream) {
    const float* data    = (const float*)d_in[0];
    const int*   indexes = (const int*)d_in[1];
    const float* W       = (const float*)d_in[2];
    const float* bias    = (const float*)d_in[3];
    float*  out = (float*)d_out;
    bf16_t* wbf = (bf16_t*)d_ws;                 // 196608 * 2 B = 384 KB

    cvt_w_kernel<<<(C_OUT * K_TOT) / 256, 256, 0, stream>>>(W, wbf);
    const int initN = B_DIM * C_OUT + B_DIM * 3 * M_NODES;
    init_out_kernel<<<(initN + 255) / 256, 256, 0, stream>>>(indexes, out);
    dim3 grid(M_NODES / MT, B_DIM);              // (16, 64)
    tree_conv_wmma_kernel<<<grid, 256, 0, stream>>>(data, indexes, wbf, bias, out);
}
